// Seq2SeqModel_52278341927194
// MI455X (gfx1250) — matrix operand order repaired and checked by reading, not compiled, not run on hardware
//
#include <hip/hip_runtime.h>

// ---------------------------------------------------------------------------
// Seq2Seq GRU encoder/decoder for MI455X (gfx1250), bf16 WMMA + f32 accumulate
// Register-blocked: each wave computes MT 16x16 m-tiles sharing B (weight)
// tiles, and the 3 GRU gates share each A tile (implicit N-blocking).
// ---------------------------------------------------------------------------

static constexpr int Bc = 256;   // batch
static constexpr int Sc = 128;   // encoder seq len
static constexpr int Tc = 64;    // decoder steps
static constexpr int Ic = 63;    // feature dim (padded to 64 for K)
static constexpr int Hc = 1024;  // hidden

typedef __attribute__((ext_vector_type(16))) __bf16 v16bf;
typedef __attribute__((ext_vector_type(8)))  float  v8f;

union BfTile {
    v16bf v;
    uint4 u[2];
};

// float -> bf16 (round to nearest even)
__device__ __forceinline__ unsigned short f2bf(float f) {
    unsigned u = __float_as_uint(f);
    unsigned r = u + 0x7FFFu + ((u >> 16) & 1u);
    return (unsigned short)(r >> 16);
}

__device__ __forceinline__ float sigmoidf_(float x) {
    return 1.0f / (1.0f + __expf(-x));
}

// A-matrix 16x32 bf16 operand, per CDNA5 ISA layout:
// lanes 0-15: M=lane, elems 0..7 = K k..k+7,  elems 8..15 = K k+16..k+23
// lanes16-31: M=lane-16, elems 0..7 = K k+8..k+15, elems 8..15 = K k+24..k+31
__device__ __forceinline__ void load_a_tile(BfTile& t, const unsigned short* row,
                                            int k, int sel) {
    const unsigned short* p = row + k + sel * 8;
    t.u[0] = *(const uint4*)(p);
    t.u[1] = *(const uint4*)(p + 16);
}

// B-matrix 32x16 bf16 operand:
// lanes 0-15: N=lane, elems 0..7 = K k..k+7,  elems 8..15 = K k+16..k+23
// lanes16-31: N=lane-16, elems 0..7 = K k+8..k+15, elems 8..15 = K k+24..k+31
__device__ __forceinline__ void load_b_tile(BfTile& t, const unsigned short* row,
                                            int k, int sel) {
    const unsigned short* p = row + k + sel * 8;
    t.u[0] = *(const uint4*)(p);
    t.u[1] = *(const uint4*)(p + 16);
}

__device__ __forceinline__ v8f wmma_bf16(const BfTile& a, const BfTile& b, v8f c) {
    return __builtin_amdgcn_wmma_f32_16x16x32_bf16(
        /*neg_a=*/false, a.v, /*neg_b=*/false, b.v,
        /*c_mod=*/(short)0, c, /*reuse_a=*/false, /*reuse_b=*/false);
}

// ---------------------------------------------------------------------------
// Utility kernels
// ---------------------------------------------------------------------------

__global__ void cvt_pad_bf16(const float* __restrict__ src,
                             unsigned short* __restrict__ dst,
                             int rows, int cols, int rows_pad, int cols_pad) {
    long i = (long)blockIdx.x * blockDim.x + threadIdx.x;
    long tot = (long)rows_pad * cols_pad;
    if (i >= tot) return;
    int r = (int)(i / cols_pad);
    int c = (int)(i % cols_pad);
    float v = (r < rows && c < cols) ? src[(long)r * cols + c] : 0.0f;
    dst[i] = f2bf(v);
}

__global__ void zero_h_kernel(float* __restrict__ hf,
                              unsigned short* __restrict__ hb, int n) {
    int i = blockIdx.x * blockDim.x + threadIdx.x;
    if (i < n) { hf[i] = 0.0f; hb[i] = 0; }
}

// ---------------------------------------------------------------------------
// Encoder GRU step, MT = 2 m-tiles per wave.
// grid = (B/(16*2), H/16), block = 32
// ---------------------------------------------------------------------------
__global__ __launch_bounds__(32)
void enc_step_kernel(const unsigned short* __restrict__ x_bf,    // [B*S, 64]
                     const unsigned short* __restrict__ wih_bf,  // [3H, 64]
                     const unsigned short* __restrict__ whh_bf,  // [3H, H]
                     const float* __restrict__ bih,
                     const float* __restrict__ bhh,
                     const unsigned short* __restrict__ h_in_bf, // [B, H]
                     const float* __restrict__ h_in_f,           // [B, H]
                     float* __restrict__ h_out_f,
                     unsigned short* __restrict__ h_out_bf,
                     int s) {
    constexpr int MT = 2;
    const int lane = threadIdx.x;
    const int n0 = blockIdx.y * 16;
    const int mn  = lane & 15;
    const int sel = lane >> 4;

    v8f acc_r[MT], acc_z[MT], acc_ni[MT], acc_nh[MT];
#pragma unroll
    for (int mi = 0; mi < MT; ++mi) {
        acc_r[mi] = (v8f){}; acc_z[mi] = (v8f){};
        acc_ni[mi] = (v8f){}; acc_nh[mi] = (v8f){};
    }

    int b0[MT];
    const unsigned short* arow[MT];
#pragma unroll
    for (int mi = 0; mi < MT; ++mi) {
        b0[mi] = (blockIdx.x * MT + mi) * 16;
        arow[mi] = h_in_bf + (size_t)(b0[mi] + mn) * Hc;
    }

    // ---- recurrent GEMM: h_in (MT*16 x H) x Whh_gate^T (H x 16), 3 gates
    const unsigned short* wr = whh_bf + (size_t)(0 * Hc + n0 + mn) * Hc;
    const unsigned short* wz = whh_bf + (size_t)(1 * Hc + n0 + mn) * Hc;
    const unsigned short* wn = whh_bf + (size_t)(2 * Hc + n0 + mn) * Hc;
    for (int k = 0; k < Hc; k += 32) {
        BfTile a[MT];
#pragma unroll
        for (int mi = 0; mi < MT; ++mi) load_a_tile(a[mi], arow[mi], k, sel);
        BfTile b1; load_b_tile(b1, wr, k, sel);
#pragma unroll
        for (int mi = 0; mi < MT; ++mi) acc_r[mi] = wmma_bf16(a[mi], b1, acc_r[mi]);
        BfTile b2; load_b_tile(b2, wz, k, sel);
#pragma unroll
        for (int mi = 0; mi < MT; ++mi) acc_z[mi] = wmma_bf16(a[mi], b2, acc_z[mi]);
        BfTile b3; load_b_tile(b3, wn, k, sel);
#pragma unroll
        for (int mi = 0; mi < MT; ++mi) acc_nh[mi] = wmma_bf16(a[mi], b3, acc_nh[mi]);
    }

    // ---- input-side GEMM (fused, K = 64 padded)
    const unsigned short* ur = wih_bf + (size_t)(0 * Hc + n0 + mn) * 64;
    const unsigned short* uz = wih_bf + (size_t)(1 * Hc + n0 + mn) * 64;
    const unsigned short* un = wih_bf + (size_t)(2 * Hc + n0 + mn) * 64;
#pragma unroll
    for (int k = 0; k < 64; k += 32) {
        BfTile a[MT];
#pragma unroll
        for (int mi = 0; mi < MT; ++mi) {
            const unsigned short* xrow =
                x_bf + ((size_t)(b0[mi] + mn) * Sc + s) * 64;
            load_a_tile(a[mi], xrow, k, sel);
        }
        BfTile b1; load_b_tile(b1, ur, k, sel);
#pragma unroll
        for (int mi = 0; mi < MT; ++mi) acc_r[mi] = wmma_bf16(a[mi], b1, acc_r[mi]);
        BfTile b2; load_b_tile(b2, uz, k, sel);
#pragma unroll
        for (int mi = 0; mi < MT; ++mi) acc_z[mi] = wmma_bf16(a[mi], b2, acc_z[mi]);
        BfTile b3; load_b_tile(b3, un, k, sel);
#pragma unroll
        for (int mi = 0; mi < MT; ++mi) acc_ni[mi] = wmma_bf16(a[mi], b3, acc_ni[mi]);
    }

    // ---- gate fusion; C layout: VGPR v -> (M = v + 8*sel, N = lane&15)
    const int Ng = n0 + mn;
    const float brr = bih[Ng] + bhh[Ng];
    const float bzz = bih[Hc + Ng] + bhh[Hc + Ng];
    const float bni = bih[2 * Hc + Ng];
    const float bnh = bhh[2 * Hc + Ng];

#pragma unroll
    for (int mi = 0; mi < MT; ++mi) {
#pragma unroll
        for (int v = 0; v < 8; ++v) {
            int row = b0[mi] + v + sel * 8;
            float r = sigmoidf_(acc_r[mi][v] + brr);
            float z = sigmoidf_(acc_z[mi][v] + bzz);
            float n = tanhf(acc_ni[mi][v] + bni + r * (acc_nh[mi][v] + bnh));
            float ho = h_in_f[(size_t)row * Hc + Ng];
            float hn = (1.0f - z) * n + z * ho;
            h_out_f[(size_t)row * Hc + Ng] = hn;
            h_out_bf[(size_t)row * Hc + Ng] = f2bf(hn);
        }
    }
}

// ---------------------------------------------------------------------------
// Decoder gh = enc_h @ dec_Whh^T + dec_bhh  (B x 3H), MT = 2.
// grid = (B/32, 3H/16), block = 32
// ---------------------------------------------------------------------------
__global__ __launch_bounds__(32)
void dec_gh_kernel(const unsigned short* __restrict__ h_bf,    // [B, H]
                   const unsigned short* __restrict__ whh_bf,  // [3H, H]
                   const float* __restrict__ bhh,
                   float* __restrict__ gh) {                   // [B, 3H]
    constexpr int MT = 2;
    const int lane = threadIdx.x;
    const int g0 = blockIdx.y * 16;
    const int mn  = lane & 15;
    const int sel = lane >> 4;

    v8f acc[MT];
    int b0[MT];
    const unsigned short* arow[MT];
#pragma unroll
    for (int mi = 0; mi < MT; ++mi) {
        acc[mi] = (v8f){};
        b0[mi] = (blockIdx.x * MT + mi) * 16;
        arow[mi] = h_bf + (size_t)(b0[mi] + mn) * Hc;
    }
    const unsigned short* brow = whh_bf + (size_t)(g0 + mn) * Hc;
    for (int k = 0; k < Hc; k += 32) {
        BfTile b; load_b_tile(b, brow, k, sel);
#pragma unroll
        for (int mi = 0; mi < MT; ++mi) {
            BfTile a; load_a_tile(a, arow[mi], k, sel);
            acc[mi] = wmma_bf16(a, b, acc[mi]);
        }
    }
    const int G = g0 + mn;
    const float bg = bhh[G];
#pragma unroll
    for (int mi = 0; mi < MT; ++mi) {
#pragma unroll
        for (int v = 0; v < 8; ++v) {
            int row = b0[mi] + v + sel * 8;
            gh[(size_t)row * (3 * Hc) + G] = acc[mi][v] + bg;
        }
    }
}

// ---------------------------------------------------------------------------
// Decoder states: every (b, t) independent. M = B*T rows, N over H.
// grid = (B*T/16, H/16), block = 32. Output bf16 (feeds projection GEMM).
// ---------------------------------------------------------------------------
__global__ __launch_bounds__(32)
void dec_state_kernel(const unsigned short* __restrict__ xdec_bf, // [B*T, 64]
                      const unsigned short* __restrict__ wih_bf,  // [3H, 64]
                      const float* __restrict__ bih,
                      const float* __restrict__ gh,               // [B, 3H]
                      const float* __restrict__ henc_f,           // [B, H]
                      unsigned short* __restrict__ state_bf) {    // [B*T, H]
    const int lane = threadIdx.x;
    const int m0 = blockIdx.x * 16;
    const int n0 = blockIdx.y * 16;
    const int mn  = lane & 15;
    const int sel = lane >> 4;

    v8f ar = {}, az = {}, ani = {};
    const unsigned short* xr = xdec_bf + (size_t)(m0 + mn) * 64;
    const unsigned short* ur = wih_bf + (size_t)(0 * Hc + n0 + mn) * 64;
    const unsigned short* uz = wih_bf + (size_t)(1 * Hc + n0 + mn) * 64;
    const unsigned short* un = wih_bf + (size_t)(2 * Hc + n0 + mn) * 64;
#pragma unroll
    for (int k = 0; k < 64; k += 32) {
        BfTile a;  load_a_tile(a, xr, k, sel);
        BfTile b1; load_b_tile(b1, ur, k, sel);
        ar = wmma_bf16(a, b1, ar);
        BfTile b2; load_b_tile(b2, uz, k, sel);
        az = wmma_bf16(a, b2, az);
        BfTile b3; load_b_tile(b3, un, k, sel);
        ani = wmma_bf16(a, b3, ani);
    }

    const int Ng = n0 + mn;
    const float bir = bih[Ng];
    const float biz = bih[Hc + Ng];
    const float bin = bih[2 * Hc + Ng];

#pragma unroll
    for (int v = 0; v < 8; ++v) {
        int row = m0 + v + sel * 8;   // flattened b*T + t
        int b = row >> 6;             // T = 64
        const float* g = gh + (size_t)b * (3 * Hc);
        float r = sigmoidf_(ar[v] + bir + g[Ng]);
        float z = sigmoidf_(az[v] + biz + g[Hc + Ng]);
        float n = tanhf(ani[v] + bin + r * g[2 * Hc + Ng]);
        float h = henc_f[(size_t)b * Hc + Ng];
        float st = (1.0f - z) * n + z * h;
        state_bf[(size_t)row * Hc + Ng] = f2bf(st);
    }
}

// ---------------------------------------------------------------------------
// Projection: out = state @ proj_W^T + proj_b. MT = 4 m-tiles per wave.
// M = B*T, K = H, N = I (padded 64). grid = (B*T/64, 4), block = 32.
// ---------------------------------------------------------------------------
__global__ __launch_bounds__(32)
void proj_kernel(const unsigned short* __restrict__ state_bf, // [B*T, H]
                 const unsigned short* __restrict__ pw_bf,    // [64, H] (row 63 zero)
                 const float* __restrict__ pb,                // [I]
                 float* __restrict__ out) {                   // [B*T, I]
    constexpr int MT = 4;
    const int lane = threadIdx.x;
    const int n0 = blockIdx.y * 16;
    const int mn  = lane & 15;
    const int sel = lane >> 4;

    v8f acc[MT];
    int m0[MT];
    const unsigned short* arow[MT];
#pragma unroll
    for (int mi = 0; mi < MT; ++mi) {
        acc[mi] = (v8f){};
        m0[mi] = (blockIdx.x * MT + mi) * 16;
        arow[mi] = state_bf + (size_t)(m0[mi] + mn) * Hc;
    }
    const unsigned short* brow = pw_bf + (size_t)(n0 + mn) * Hc;
    for (int k = 0; k < Hc; k += 32) {
        BfTile b; load_b_tile(b, brow, k, sel);
#pragma unroll
        for (int mi = 0; mi < MT; ++mi) {
            BfTile a; load_a_tile(a, arow[mi], k, sel);
            acc[mi] = wmma_bf16(a, b, acc[mi]);
        }
    }
    const int O = n0 + mn;
    const float bo = (O < Ic) ? pb[O] : 0.0f;
#pragma unroll
    for (int mi = 0; mi < MT; ++mi) {
#pragma unroll
        for (int v = 0; v < 8; ++v) {
            int row = m0[mi] + v + sel * 8;
            if (O < Ic) out[(size_t)row * Ic + O] = acc[mi][v] + bo;
        }
    }
}

// ---------------------------------------------------------------------------
// Host-side orchestration
// ---------------------------------------------------------------------------
extern "C" void kernel_launch(void* const* d_in, const int* in_sizes, int n_in,
                              void* d_out, int out_size, void* d_ws, size_t ws_size,
                              hipStream_t stream) {
    (void)in_sizes; (void)n_in; (void)out_size; (void)ws_size;

    const float* enc_x   = (const float*)d_in[0];
    const float* dec_x   = (const float*)d_in[1];
    const float* enc_Wih = (const float*)d_in[2];
    const float* enc_Whh = (const float*)d_in[3];
    const float* enc_bih = (const float*)d_in[4];
    const float* enc_bhh = (const float*)d_in[5];
    const float* dec_Wih = (const float*)d_in[6];
    const float* dec_Whh = (const float*)d_in[7];
    const float* dec_bih = (const float*)d_in[8];
    const float* dec_bhh = (const float*)d_in[9];
    const float* proj_W  = (const float*)d_in[10];
    const float* proj_b  = (const float*)d_in[11];
    float* out = (float*)d_out;

    char* ws = (char*)d_ws;
    size_t off = 0;
    auto take = [&](size_t bytes) -> char* {
        char* p = ws + off;
        off += (bytes + 255) & ~(size_t)255;
        return p;
    };
    unsigned short* whh_enc_bf = (unsigned short*)take((size_t)3 * Hc * Hc * 2);
    unsigned short* whh_dec_bf = (unsigned short*)take((size_t)3 * Hc * Hc * 2);
    unsigned short* wih_enc_bf = (unsigned short*)take((size_t)3 * Hc * 64 * 2);
    unsigned short* wih_dec_bf = (unsigned short*)take((size_t)3 * Hc * 64 * 2);
    unsigned short* projw_bf   = (unsigned short*)take((size_t)64 * Hc * 2);
    unsigned short* xenc_bf    = (unsigned short*)take((size_t)Bc * Sc * 64 * 2);
    unsigned short* xdec_bf    = (unsigned short*)take((size_t)Bc * Tc * 64 * 2);
    float*          h0f        = (float*)take((size_t)Bc * Hc * 4);
    float*          h1f        = (float*)take((size_t)Bc * Hc * 4);
    unsigned short* h0b        = (unsigned short*)take((size_t)Bc * Hc * 2);
    unsigned short* h1b        = (unsigned short*)take((size_t)Bc * Hc * 2);
    float*          ghd        = (float*)take((size_t)Bc * 3 * Hc * 4);
    unsigned short* state_bf   = (unsigned short*)take((size_t)Bc * Tc * Hc * 2);

    auto cvt = [&](const float* s, unsigned short* d, long rows, long cols,
                   long rp, long cp) {
        long tot = rp * cp;
        int blocks = (int)((tot + 255) / 256);
        hipLaunchKernelGGL(cvt_pad_bf16, dim3(blocks), dim3(256), 0, stream,
                           s, d, (int)rows, (int)cols, (int)rp, (int)cp);
    };

    // Weight / input conversion (once per call; bf16 weights live in L2)
    cvt(enc_Whh, whh_enc_bf, 3 * Hc, Hc, 3 * Hc, Hc);
    cvt(dec_Whh, whh_dec_bf, 3 * Hc, Hc, 3 * Hc, Hc);
    cvt(enc_Wih, wih_enc_bf, 3 * Hc, Ic, 3 * Hc, 64);
    cvt(dec_Wih, wih_dec_bf, 3 * Hc, Ic, 3 * Hc, 64);
    cvt(proj_W,  projw_bf,   Ic,     Hc, 64,     Hc);
    cvt(enc_x,   xenc_bf,    (long)Bc * Sc, Ic, (long)Bc * Sc, 64);
    cvt(dec_x,   xdec_bf,    (long)Bc * Tc, Ic, (long)Bc * Tc, 64);

    // h0 = 0
    {
        int n = Bc * Hc;
        hipLaunchKernelGGL(zero_h_kernel, dim3((n + 255) / 256), dim3(256), 0,
                           stream, h0f, h0b, n);
    }

    // ---- encoder recurrence: 128 stream-ordered steps, ping-pong h buffers
    float*          hF[2] = {h0f, h1f};
    unsigned short* hB[2] = {h0b, h1b};
    for (int s = 0; s < Sc; ++s) {
        int i = s & 1, o = i ^ 1;
        hipLaunchKernelGGL(enc_step_kernel, dim3(Bc / 32, Hc / 16), dim3(32), 0,
                           stream, xenc_bf, wih_enc_bf, whh_enc_bf, enc_bih,
                           enc_bhh, hB[i], hF[i], hF[o], hB[o], s);
    }
    // Sc = 128 (even) -> final hidden state back in buffer 0
    float* henc_f = hF[0];
    unsigned short* henc_b = hB[0];

    // ---- decoder hidden-side gates (one GEMM, reused for all T steps)
    hipLaunchKernelGGL(dec_gh_kernel, dim3(Bc / 32, 3 * Hc / 16), dim3(32), 0,
                       stream, henc_b, whh_dec_bf, dec_bhh, ghd);

    // ---- decoder states for all (b, t) at once
    hipLaunchKernelGGL(dec_state_kernel, dim3(Bc * Tc / 16, Hc / 16), dim3(32), 0,
                       stream, xdec_bf, wih_dec_bf, dec_bih, ghd, henc_f,
                       state_bf);

    // ---- projection to outputs (B*T, I)
    hipLaunchKernelGGL(proj_kernel, dim3(Bc * Tc / 64, 64 / 16), dim3(32), 0,
                       stream, state_bf, projw_bf, proj_b, out);
}
